// GCNEncoder_2491081031685
// MI455X (gfx1250) — compile-verified, matrix operand order checked by reading, not run on hardware
//
#include <hip/hip_runtime.h>
#include <hip/hip_bf16.h>

typedef float v2f  __attribute__((ext_vector_type(2)));
typedef float v8f  __attribute__((ext_vector_type(8)));
typedef __bf16 v16bf __attribute__((ext_vector_type(16)));

#define CH 128

// Native non-returning f32 atomic add, device scope (RMW resolves in L2).
// Guarantees global_atomic_add_f32 instead of a CAS expansion loop.
__device__ __forceinline__ void atomic_add_f32_dev(float* addr, float val) {
    asm volatile("global_atomic_add_f32 %0, %1, off scope:SCOPE_DEV"
                 :: "v"(addr), "v"(val)
                 : "memory");
}

// ---------------- zero fill ------------------------------------------------
__global__ __launch_bounds__(256) void gcn_zero_kernel(float* __restrict__ p, long n) {
    long i = (long)blockIdx.x * blockDim.x + threadIdx.x;
    long stride = (long)gridDim.x * blockDim.x;
    for (long j = i * 4; j + 3 < n; j += stride * 4) {
        *(float4*)(p + j) = make_float4(0.f, 0.f, 0.f, 0.f);
    }
}

// ---------------- degree: deg[dst] += 1 ------------------------------------
__global__ __launch_bounds__(256) void gcn_deg_kernel(const int* __restrict__ dst,
                                                      float* __restrict__ deg, int E) {
    int e = blockIdx.x * blockDim.x + threadIdx.x;
    if (e < E) {
        atomic_add_f32_dev(&deg[dst[e]], 1.0f);
    }
}

// ---------------- dinv[i] = rsqrt(deg[i] + 1) (self loop) ------------------
__global__ __launch_bounds__(256) void gcn_dinv_kernel(float* __restrict__ deg, int N) {
    int i = blockIdx.x * blockDim.x + threadIdx.x;
    if (i < N) deg[i] = rsqrtf(deg[i] + 1.0f);
}

// ---------------- coef[e] = dinv[src[e]] * dinv[dst[e]] --------------------
__global__ __launch_bounds__(256) void gcn_coef_kernel(const int* __restrict__ src,
                                                       const int* __restrict__ dst,
                                                       const float* __restrict__ dinv,
                                                       float* __restrict__ coef, int E) {
    int e = blockIdx.x * blockDim.x + threadIdx.x;
    if (e < E) coef[e] = dinv[src[e]] * dinv[dst[e]];
}

// ---------------- H = X @ W via WMMA ---------------------------------------
// One wave computes a 16x16 output tile. Block = 256 threads = 8 waves
// covering all 128 output columns of a 16-row stripe. nrows % 16 == 0.
__global__ __launch_bounds__(256) void gcn_gemm_kernel(const float* __restrict__ X,
                                                       const float* __restrict__ W,
                                                       float* __restrict__ H) {
    const int wave = threadIdx.x >> 5;
    const int lane = threadIdx.x & 31;
    const int row0 = blockIdx.x * 16;
    const int col0 = wave * 16;
    const int m    = lane & 15;          // M for A-frag, N for B-frag
    const int hi   = lane >> 4;          // 0: lanes 0-15, 1: lanes 16-31

    const float* __restrict__ xr = X + (long)(row0 + m) * CH;  // A row (M = m)
    const float* __restrict__ wc = W + (col0 + m);             // B column (N = m)

    v8f acc = {};
#if __has_builtin(__builtin_amdgcn_wmma_f32_16x16x4_f32)
    const int khalf = hi * 2;            // lanes 0-15: K0,K1 ; lanes 16-31: K2,K3
    for (int k = 0; k < CH; k += 4) {
        v2f a, b;
        a.x = xr[k + khalf];
        a.y = xr[k + khalf + 1];
        b.x = wc[(k + khalf) * CH];
        b.y = wc[(k + khalf + 1) * CH];
        acc = __builtin_amdgcn_wmma_f32_16x16x4_f32(
            /*neg_a=*/false, a, /*neg_b=*/false, b,
            /*c_mod=*/(short)0, acc, /*reuse_a=*/false, /*reuse_b=*/false);
    }
#else
    // Fallback: bf16 16x16x32 WMMA (confirmed builtin). Slight precision loss.
    const int kb = hi * 8;
    for (int k = 0; k < CH; k += 32) {
        v16bf a, b;
        #pragma unroll
        for (int j = 0; j < 8; ++j) {
            a[j]     = (__bf16)xr[k + kb + j];
            a[8 + j] = (__bf16)xr[k + 16 + kb + j];
            b[j]     = (__bf16)wc[(k + kb + j) * CH];
            b[8 + j] = (__bf16)wc[(k + 16 + kb + j) * CH];
        }
        acc = __builtin_amdgcn_wmma_f32_16x16x32_bf16(
            false, a, false, b, (short)0, acc, false, false);
    }
#endif
    // C/D layout: VGPR r -> M = r (lanes 0-15) / M = 8+r (lanes 16-31), N = lane&15
    const int mbase = row0 + hi * 8;
    #pragma unroll
    for (int r = 0; r < 8; ++r) {
        H[(long)(mbase + r) * CH + col0 + m] = acc[r];
    }
}

// ---------------- edge scatter: agg[dst] += h[src] * coef[e] ---------------
// One wave per edge; each lane handles 4 consecutive channels (32*4 = 128).
__global__ __launch_bounds__(256) void gcn_scatter_kernel(const int* __restrict__ src,
                                                          const int* __restrict__ dst,
                                                          const float* __restrict__ h,
                                                          const float* __restrict__ coef,
                                                          float* __restrict__ agg, int E) {
    int gid  = blockIdx.x * blockDim.x + threadIdx.x;
    int e    = gid >> 5;
    int lane = threadIdx.x & 31;
    if (e >= E) return;
    int s = src[e];
    int d = dst[e];
    float c = coef[e];
    float4 v = *(const float4*)(h + (long)s * CH + lane * 4);
    float* ad = agg + (long)d * CH + lane * 4;
    atomic_add_f32_dev(ad + 0, v.x * c);
    atomic_add_f32_dev(ad + 1, v.y * c);
    atomic_add_f32_dev(ad + 2, v.z * c);
    atomic_add_f32_dev(ad + 3, v.w * c);
}

// ---------------- finalize: out = relu(agg + h * dinv^2 + b) (in place ok) --
// float4-vectorized: 32 quads per node row.
__global__ __launch_bounds__(256) void gcn_finalize_kernel(float4* __restrict__ out,
                                                           const float4* __restrict__ agg,
                                                           const float4* __restrict__ h,
                                                           const float* __restrict__ dinv,
                                                           const float4* __restrict__ bias,
                                                           long nquads) {
    long idx = (long)blockIdx.x * blockDim.x + threadIdx.x;
    if (idx >= nquads) return;
    int i = (int)(idx >> 5);         // node (CH/4 = 32 quads per node)
    int c = (int)(idx & 31);         // quad within row
    float di = dinv[i];
    float sl = di * di;              // self-loop norm
    float4 a  = agg[idx];
    float4 hv = h[idx];
    float4 b  = bias[c];
    float4 r;
    r.x = fmaxf(fmaf(hv.x, sl, a.x) + b.x, 0.0f);
    r.y = fmaxf(fmaf(hv.y, sl, a.y) + b.y, 0.0f);
    r.z = fmaxf(fmaf(hv.z, sl, a.z) + b.z, 0.0f);
    r.w = fmaxf(fmaf(hv.w, sl, a.w) + b.w, 0.0f);
    out[idx] = r;
}

extern "C" void kernel_launch(void* const* d_in, const int* in_sizes, int n_in,
                              void* d_out, int out_size, void* d_ws, size_t ws_size,
                              hipStream_t stream) {
    const float* x  = (const float*)d_in[0];
    const int*   ei = (const int*)d_in[1];
    const float* W1 = (const float*)d_in[2];
    const float* b1 = (const float*)d_in[3];
    const float* W2 = (const float*)d_in[4];
    const float* b2 = (const float*)d_in[5];

    const int N = in_sizes[0] / CH;       // 100000 (multiple of 16)
    const int E = in_sizes[1] / 2;        // 1600000
    const int* src = ei;
    const int* dst = ei + E;

    // Workspace: dinv[N] | coef[E] | bufH[N*128] | bufA[N*128]  (~109 MB)
    float* ws   = (float*)d_ws;
    float* dinv = ws;
    float* coef = dinv + (((long)N + 255) & ~255L);
    float* bufH = coef + (((long)E + 255) & ~255L);
    float* bufA = bufH + (long)N * CH;
    float* out  = (float*)d_out;

    const long feat   = (long)N * CH;
    const long nquads = feat / 4;
    const int  TB     = 256;
    const int  gE     = (E + TB - 1) / TB;
    const int  gN     = (N + TB - 1) / TB;
    const int  gEdge  = (E * 32 + TB - 1) / TB;       // one wave per edge
    const int  gQuad  = (int)((nquads + TB - 1) / TB);
    const int  gZero  = 2048;                          // grid-stride float4 zero
    const int  gGemm  = N / 16;

    // degrees -> dinv -> per-edge coef (recomputed every call; stateless)
    gcn_zero_kernel<<<gZero, TB, 0, stream>>>(dinv, (long)N);
    gcn_deg_kernel<<<gE, TB, 0, stream>>>(dst, dinv, E);
    gcn_dinv_kernel<<<gN, TB, 0, stream>>>(dinv, N);
    gcn_coef_kernel<<<gE, TB, 0, stream>>>(src, dst, dinv, coef, E);

    // ---- layer 1 ----
    gcn_gemm_kernel<<<gGemm, TB, 0, stream>>>(x, W1, bufH);
    gcn_zero_kernel<<<gZero, TB, 0, stream>>>(bufA, feat);
    gcn_scatter_kernel<<<gEdge, TB, 0, stream>>>(src, dst, bufH, coef, bufA, E);
    gcn_finalize_kernel<<<gQuad, TB, 0, stream>>>((float4*)bufA, (const float4*)bufA,
                                                  (const float4*)bufH, dinv,
                                                  (const float4*)b1, nquads);

    // ---- layer 2 ---- (aggregate directly into d_out)
    gcn_gemm_kernel<<<gGemm, TB, 0, stream>>>(bufA, W2, bufH);
    gcn_zero_kernel<<<gZero, TB, 0, stream>>>(out, feat);
    gcn_scatter_kernel<<<gEdge, TB, 0, stream>>>(src, dst, bufH, coef, out, E);
    gcn_finalize_kernel<<<gQuad, TB, 0, stream>>>((float4*)out, (const float4*)out,
                                                  (const float4*)bufH, dinv,
                                                  (const float4*)b2, nquads);
}
